// GNNEncoder_39779987095911
// MI455X (gfx1250) — matrix operand order repaired and checked by reading, compiled, not run on hardware
//
#include <hip/hip_runtime.h>
#include <hip/hip_bf16.h>
#include <cstdint>

#define D 128  // feature dim (both d_in and d_h are 128)

typedef __attribute__((ext_vector_type(16))) __bf16 v16bf;
typedef __attribute__((ext_vector_type(8)))  float  v8f;

__device__ __forceinline__ __bf16 f2bf(float f) {
    unsigned u = __builtin_bit_cast(unsigned, f);
    // round-to-nearest-even bf16 truncation
    unsigned r = (u + 0x7FFFu + ((u >> 16) & 1u)) >> 16;
    unsigned short s = (unsigned short)r;
    return __builtin_bit_cast(__bf16, s);
}

__device__ __forceinline__ void atomAddF(float* p, float v) {
    __hip_atomic_fetch_add(p, v, __ATOMIC_RELAXED, __HIP_MEMORY_SCOPE_AGENT);
}

// ---- degree / normalization -------------------------------------------------
__global__ void deg_init_kernel(float* dis, int n) {
    int i = blockIdx.x * blockDim.x + threadIdx.x;
    if (i < n) dis[i] = 1.0f;  // self-loop contributes 1 to degree
}

__global__ void deg_edge_kernel(const long long* __restrict__ col, float* dis, int e) {
    int i = blockIdx.x * blockDim.x + threadIdx.x;
    if (i < e) atomAddF(&dis[(int)col[i]], 1.0f);
}

__global__ void deg_finish_kernel(float* dis, int n) {
    int i = blockIdx.x * blockDim.x + threadIdx.x;
    if (i < n) {
        float d = dis[i];
        dis[i] = (d > 0.0f) ? rsqrtf(d) : 0.0f;
    }
}

// ---- fp32 -> bf16 conversion (with optional fused ReLU), zero-pads tail ----
__global__ void cvt_x_kernel(const float* __restrict__ src, __bf16* __restrict__ dst,
                             int nvalid, int ntotal, int relu) {
    int i = blockIdx.x * blockDim.x + threadIdx.x;
    if (i >= ntotal) return;
    float v = (i < nvalid) ? src[i] : 0.0f;
    if (relu) v = fmaxf(v, 0.0f);
    dst[i] = f2bf(v);
}

// ---- W (K x N row-major) -> bf16 transposed (N x K) for contiguous B frags --
__global__ void cvt_w_kernel(const float* __restrict__ w, __bf16* __restrict__ wt) {
    int i = blockIdx.x * blockDim.x + threadIdx.x;  // i = n*D + k
    if (i >= D * D) return;
    int n = i >> 7, k = i & (D - 1);
    wt[i] = f2bf(w[k * D + n]);
}

// ---- WMMA GEMM: H[NPAD x 128] = A[NPAD x 128] (bf16) * W[128 x 128] --------
// block = 256 threads = 8 waves; each wave computes a 16-row x 128-col strip.
// sched_barrier pins all K-slice fragment loads before the 8-WMMA burst so the
// XDL pipe issues back-to-back after a single ds wait.
__global__ void __launch_bounds__(256) gemm_kernel(const __bf16* __restrict__ A,
                                                   const __bf16* __restrict__ Wt,
                                                   float* __restrict__ H) {
    __shared__ __bf16 Wl[D * D];  // transposed W: Wl[n*D + k], 32 KB
    {
        const unsigned* s = (const unsigned*)Wt;
        unsigned* d = (unsigned*)Wl;
        for (int i = threadIdx.x; i < D * D / 2; i += 256) d[i] = s[i];
    }
    __syncthreads();

    const int wave  = threadIdx.x >> 5;
    const int lane  = threadIdx.x & 31;
    const int row0  = blockIdx.x * 128 + wave * 16;
    const int m     = lane & 15;
    const int khalf = (lane < 16) ? 0 : 16;  // A-matrix K sub-block select
    const int kbase = (lane < 16) ? 0 : 16;  // B-matrix K sub-block select

    v8f acc[8];
#pragma unroll
    for (int t = 0; t < 8; ++t) acc[t] = (v8f){0.f,0.f,0.f,0.f,0.f,0.f,0.f,0.f};

#pragma unroll
    for (int kk = 0; kk < D; kk += 32) {
        // A fragment (16x32 bf16): lane holds row m, one contiguous 16-elem chunk
        v16bf a;
        const __bf16* ap = A + (size_t)(row0 + m) * D + kk + khalf;
#pragma unroll
        for (int e2 = 0; e2 < 8; ++e2) { a[e2] = ap[e2]; a[8 + e2] = ap[8 + e2]; }

        // Preload all 8 B fragments of this K-slice (16 contiguous bf16 each)
        v16bf b[8];
#pragma unroll
        for (int t = 0; t < 8; ++t) {
            const __bf16* bp = &Wl[(t * 16 + m) * D + kk + kbase];
#pragma unroll
            for (int e2 = 0; e2 < 16; ++e2) b[t][e2] = bp[e2];
        }

        // Keep the loads above, the WMMA burst below.
        __builtin_amdgcn_sched_barrier(0);

        // Fire the 8 WMMAs back-to-back
#pragma unroll
        for (int t = 0; t < 8; ++t) {
            acc[t] = __builtin_amdgcn_wmma_f32_16x16x32_bf16(
                false, a, false, b[t], (short)0, acc[t], false, false);
        }
    }

    // D-matrix layout: element r of acc -> row = (lane<16 ? r : 8+r), col = t*16 + (lane&15)
#pragma unroll
    for (int t = 0; t < 8; ++t) {
        const int n = t * 16 + m;
#pragma unroll
        for (int r = 0; r < 8; ++r) {
            const int rr = row0 + ((lane < 16) ? r : (8 + r));
            H[(size_t)rr * D + n] = acc[t][r];
        }
    }
}

// ---- out = dis[i]^2 * H[i] + bias  (self-loop term + bias) ------------------
__global__ void init_out_kernel(const float* __restrict__ H, const float* __restrict__ dis,
                                const float* __restrict__ bias, float* __restrict__ out,
                                int n) {
    int i = blockIdx.x * blockDim.x + threadIdx.x;
    if (i >= n * D) return;
    int node = i >> 7, f = i & (D - 1);
    float s = dis[node];
    out[i] = s * s * H[i] + bias[f];
}

// ---- edge scatter-add: one wave per edge, one float4 per lane ---------------
__global__ void edge_agg_kernel(const long long* __restrict__ row,
                                const long long* __restrict__ col,
                                const float* __restrict__ dis,
                                const float* __restrict__ H,
                                float* __restrict__ out, int e) {
    int gid  = blockIdx.x * blockDim.x + threadIdx.x;
    int eid  = gid >> 5;
    int lane = gid & 31;
    if (eid >= e) return;
    int r = (int)row[eid];
    int c = (int)col[eid];
    float nrm = dis[r] * dis[c];
    const float4 v = ((const float4*)(H + (size_t)r * D))[lane];
    float* o = out + (size_t)c * D + lane * 4;
    atomAddF(o + 0, nrm * v.x);
    atomAddF(o + 1, nrm * v.y);
    atomAddF(o + 2, nrm * v.z);
    atomAddF(o + 3, nrm * v.w);
}

__global__ void relu_kernel(float* out, int n) {
    int i = blockIdx.x * blockDim.x + threadIdx.x;
    if (i < n) out[i] = fmaxf(out[i], 0.0f);
}

extern "C" void kernel_launch(void* const* d_in, const int* in_sizes, int n_in,
                              void* d_out, int out_size, void* d_ws, size_t ws_size,
                              hipStream_t stream) {
    const float*     x  = (const float*)d_in[0];
    const long long* ei = (const long long*)d_in[1];
    const float* Ws[3] = {(const float*)d_in[2], (const float*)d_in[4], (const float*)d_in[6]};
    const float* Bs[3] = {(const float*)d_in[3], (const float*)d_in[5], (const float*)d_in[7]};
    float* out = (float*)d_out;

    const int N    = in_sizes[0] / D;      // 50000
    const int E    = in_sizes[1] / 2;      // 640000
    const int NPAD = ((N + 127) / 128) * 128;

    const long long* erow = ei;
    const long long* ecol = ei + E;

    // workspace layout (256B aligned blocks)
    uint8_t* ws = (uint8_t*)d_ws;
    size_t off = 0;
    float*  dis = (float*)(ws + off);  off += (((size_t)N * 4) + 255) & ~(size_t)255;
    __bf16* Abf = (__bf16*)(ws + off); off += (((size_t)NPAD * D * 2) + 255) & ~(size_t)255;
    __bf16* Wbf = (__bf16*)(ws + off); off += (((size_t)D * D * 2) + 255) & ~(size_t)255;
    float*  H   = (float*)(ws + off);

    const int TB = 256;

    // symmetric normalization coefficients (shared by all 3 layers)
    deg_init_kernel  <<<(N + TB - 1) / TB, TB, 0, stream>>>(dis, N);
    deg_edge_kernel  <<<(E + TB - 1) / TB, TB, 0, stream>>>(ecol, dis, E);
    deg_finish_kernel<<<(N + TB - 1) / TB, TB, 0, stream>>>(dis, N);

    for (int L = 0; L < 3; ++L) {
        const float* src = (L == 0) ? x : out;  // ReLU of prev layer fused into cvt
        cvt_x_kernel<<<((size_t)NPAD * D + TB - 1) / TB, TB, 0, stream>>>(
            src, Abf, N * D, NPAD * D, L > 0 ? 1 : 0);
        cvt_w_kernel<<<(D * D + TB - 1) / TB, TB, 0, stream>>>(Ws[L], Wbf);
        gemm_kernel<<<NPAD / 128, 256, 0, stream>>>(Abf, Wbf, H);
        init_out_kernel<<<((size_t)N * D + TB - 1) / TB, TB, 0, stream>>>(H, dis, Bs[L], out, N);
        edge_agg_kernel<<<(((size_t)E * 32) + TB - 1) / TB, TB, 0, stream>>>(
            erow, ecol, dis, H, out, E);
    }
    relu_kernel<<<((size_t)N * D + TB - 1) / TB, TB, 0, stream>>>(out, N * D);
}
